// DistillKL_59356448031310
// MI455X (gfx1250) — matrix operand order repaired and checked by reading, not compiled, not run on hardware
//
#include <hip/hip_runtime.h>
#include <math.h>

typedef __attribute__((ext_vector_type(16))) _Float16 v16h;
typedef __attribute__((ext_vector_type(8)))  float    v8f;

#define BSZ 64
#define C   1000
#define CP  1024      // padded length (multiple of 32 for WMMA k-chunks)
#define NTHREADS 256  // 8 wave32

// One block per batch row. Computes rowsum_b = sum_i pt*log(ps) -> ws[b].
__global__ __launch_bounds__(NTHREADS)
void distillkl_row_kernel(const float* __restrict__ ys,
                          const float* __restrict__ yt,
                          const float* __restrict__ wf,
                          float* __restrict__ rowsums) {
    __shared__ float    s_xs[CP];
    __shared__ float    s_xt[CP];
    __shared__ float    s_w [CP];
    __shared__ _Float16 s_wh[CP];
    __shared__ float    s_t3s[CP];
    __shared__ float    s_t3t[CP];
    __shared__ float    s_red[NTHREADS];
    __shared__ float    s_red2[NTHREADS];
    __shared__ float    s_negs, s_negt;

    const int b   = blockIdx.x;
    const int tid = threadIdx.x;
    const float* ysr = ys + b * C;
    const float* ytr = yt + b * C;
    const float* wr  = wf + b * C;

    // ---- Stage rows into LDS (x = y/4), zero-pad to CP ----
    for (int i = tid; i < CP; i += NTHREADS) {
        float vs = 0.0f, vt = 0.0f, vw = 0.0f;
        if (i < C) { vs = ysr[i] * 0.25f; vt = ytr[i] * 0.25f; vw = wr[i]; }
        s_xs[i] = vs; s_xt[i] = vt; s_w[i] = vw;
        s_wh[i] = (_Float16)vw;   // padding w=0 annihilates padded k in WMMA
    }
    __syncthreads();

    // ---- neg_sum_{s,t} = sum_k (1-w)*exp(x) ----
    float ns = 0.0f, nt = 0.0f;
    for (int i = tid; i < C; i += NTHREADS) {
        float om = 1.0f - s_w[i];
        ns += om * __expf(s_xs[i]);
        nt += om * __expf(s_xt[i]);
    }
    s_red[tid] = ns; s_red2[tid] = nt;
    __syncthreads();
    for (int off = NTHREADS / 2; off > 0; off >>= 1) {
        if (tid < off) { s_red[tid] += s_red[tid + off]; s_red2[tid] += s_red2[tid + off]; }
        __syncthreads();
    }
    if (tid == 0) { s_negs = s_red[0]; s_negt = s_red2[0]; }
    __syncthreads();

    // ---- t3 via WMMA f32_16x16x32_f16: D(16x16) += ReluDiff(16x32) x Wbcast(32x16)
    //      Student & teacher fused per k-chunk: B (broadcast w) loaded once,
    //      two independent WMMAs issued back-to-back (no RAW hazard). ----
    const int wave  = tid >> 5;
    const int lane  = tid & 31;
    const int halfw = lane >> 4;        // 0: lanes 0-15, 1: lanes 16-31
    const int rlane = lane & 15;        // matrix row within tile
    const int koffA = halfw * 8;        // A-fragment K base per ISA layout
    const int bofsB = halfw * 16;       // unused: B is read with koffA, same k order as A

    for (int tile = wave; tile < CP / 16; tile += NTHREADS / 32) {
        const int rowbase = tile * 16;
        const float xis = s_xs[rowbase + rlane];
        const float xit = s_xt[rowbase + rlane];
        v8f accs = {};
        v8f acct = {};
        for (int kc = 0; kc < CP; kc += 32) {
            v16h bm;
            #pragma unroll
            for (int j = 0; j < 16; ++j) bm[j] = s_wh[kc + koffA + (j & 7) + ((j >> 3) << 4)];
            v16h as, at;
            #pragma unroll
            for (int j = 0; j < 8; ++j) {
                const float x0s = s_xs[kc + koffA + j];
                const float x1s = s_xs[kc + 16 + koffA + j];
                const float x0t = s_xt[kc + koffA + j];
                const float x1t = s_xt[kc + 16 + koffA + j];
                as[j]     = (_Float16)fmaxf(xis - x0s, 0.0f);
                as[8 + j] = (_Float16)fmaxf(xis - x1s, 0.0f);
                at[j]     = (_Float16)fmaxf(xit - x0t, 0.0f);
                at[8 + j] = (_Float16)fmaxf(xit - x1t, 0.0f);
            }
            // D = A*B + C  (8 args: neg_a, A, neg_b, B, c_mod, C, reuse_a, reuse_b)
            accs = __builtin_amdgcn_wmma_f32_16x16x32_f16(
                       false, as, false, bm, (short)0, accs, false, false);
            acct = __builtin_amdgcn_wmma_f32_16x16x32_f16(
                       false, at, false, bm, (short)0, acct, false, false);
        }
        // D columns are identical (B columns identical); lanes 0 and 16
        // hold rows 0-7 / 8-15 of the tile in acc[0..7].
        if (rlane == 0) {
            const int rb = rowbase + halfw * 8;
            #pragma unroll
            for (int r = 0; r < 8; ++r) {
                s_t3s[rb + r] = accs[r] * (1.0f / (float)C);
                s_t3t[rb + r] = acct[r] * (1.0f / (float)C);
            }
        }
    }
    __syncthreads();

    // ---- Pointwise loss: sum_i pt * log(ps) ----
    float part = 0.0f;
    const float negs = s_negs, negt = s_negt;
    for (int i = tid; i < C; i += NTHREADS) {
        const float wv  = s_w[i];
        const float omw = 1.0f - wv;
        const float exs = __expf(s_xs[i]);
        const float exv = __expf(s_xt[i]);
        const float chs = wv * __expf(-s_t3s[i]) + omw;
        const float cht = wv * __expf(-s_t3t[i]) + omw;
        const float ps  = exs / (chs * negs + exs);
        const float pt  = exv / (cht * negt + exv);
        const float ps1 = fminf(ps, 1.0f);
        const float pt1 = fminf(pt, 1.0f);
        const float zp  = fminf(fabsf((pt1 + 1.0f) * 0.5f - ps1), 1.0f);
        const float pos = wv * sqrtf(sqrtf(zp));           // z^0.25
        const float zn  = fminf(fabsf(pt1 * 0.5f - ps1), 1.0f);
        const float ex  = pos + omw * zn;                  // expt
        const float psE = fmaxf(__expf(ex * __logf(ps1)), 1e-7f);  // ps1^expt
        const float ptv = fmaxf(pt, 1e-7f);
        part += ptv * __logf(psE);
    }
    s_red[tid] = part;
    __syncthreads();
    for (int off = NTHREADS / 2; off > 0; off >>= 1) {
        if (tid < off) s_red[tid] += s_red[tid + off];
        __syncthreads();
    }
    if (tid == 0) rowsums[b] = s_red[0];
}

// Deterministic finalize: loss = -(T*T/B) * sum_b rowsum_b = -0.25 * sum
__global__ void distillkl_finalize_kernel(const float* __restrict__ rowsums,
                                          float* __restrict__ out) {
    __shared__ float r[BSZ];
    const int t = threadIdx.x;
    r[t] = rowsums[t];
    __syncthreads();
    for (int off = BSZ / 2; off > 0; off >>= 1) {
        if (t < off) r[t] += r[t + off];
        __syncthreads();
    }
    if (t == 0) out[0] = -0.25f * r[0];
}

extern "C" void kernel_launch(void* const* d_in, const int* in_sizes, int n_in,
                              void* d_out, int out_size, void* d_ws, size_t ws_size,
                              hipStream_t stream) {
    (void)in_sizes; (void)n_in; (void)out_size; (void)ws_size;
    const float* ys = (const float*)d_in[0];
    const float* yt = (const float*)d_in[1];
    const float* wf = (const float*)d_in[2];
    float* rowsums  = (float*)d_ws;          // 64 floats of scratch
    float* out      = (float*)d_out;

    distillkl_row_kernel<<<BSZ, NTHREADS, 0, stream>>>(ys, yt, wf, rowsums);
    distillkl_finalize_kernel<<<1, BSZ, 0, stream>>>(rowsums, out);
}
